// SSBlock_61933428417262
// MI455X (gfx1250) — matrix operand order repaired and checked by reading, not compiled, not run on hardware
//
#include <hip/hip_runtime.h>
#include <hip/hip_bf16.h>
#include <math.h>

typedef __bf16 bf16;
typedef __attribute__((ext_vector_type(16))) __bf16 v16bf;
typedef __attribute__((ext_vector_type(8)))  __bf16 v8bf;
typedef __attribute__((ext_vector_type(8)))  float  v8f;

#define DEV __device__ __forceinline__

static constexpr int BATCH = 8, CH = 64, IMG = 128, FS = 4, NPIX = 1024, EMB = 64;

// ---- attention LDS layout (elements of bf16) ----
static constexpr int QSTR = 72;    // 1024 x 72  (row = n, col = c/e)
static constexpr int VSTR = 1032;  // 64 x 1032  (row = c, col = key)
static constexpr int WSTR = 72;    // 64 x 72    (row = e, col = c)
static constexpr int PSTR = 40;    // per-wave 16 x 40 staging
static constexpr int ATT_V_OFF = NPIX * QSTR;
static constexpr int ATT_W_OFF = ATT_V_OFF + CH * VSTR;
static constexpr int ATT_P_OFF = ATT_W_OFF + EMB * WSTR;
static constexpr int ATT_ONE_OFF = ATT_P_OFF + 8 * 16 * PSTR;
static constexpr int ATT_LDS_ELEMS = ATT_ONE_OFF + 16 * 32;   // + ones tile
static constexpr size_t ATT_LDS_BYTES = (size_t)ATT_LDS_ELEMS * sizeof(bf16);  // 300032

// ---- conv3x3 LDS layout ----
static constexpr int PATCH = 36, CSTR = 72, WCSTR = 584;
static constexpr int CV_W_OFF = PATCH * PATCH * CSTR;
static constexpr int CV_LDS_ELEMS = CV_W_OFF + CH * WCSTR;
static constexpr size_t CV_LDS_BYTES = (size_t)CV_LDS_ELEMS * sizeof(bf16);    // 261376

// ---- conv1x1 LDS layout ----
static constexpr int K3_W_OFF = NPIX * QSTR;
static constexpr int K3_LDS_ELEMS = K3_W_OFF + CH * WSTR;
static constexpr size_t K3_LDS_BYTES = (size_t)K3_LDS_ELEMS * sizeof(bf16);    // 156672

static constexpr float SOFTMAX_SCL = 0.125f * 1.44269504088896340736f;  // (1/sqrt(E))*log2(e)

// ---------------- fragment helpers (ISA §7.12.2 layouts, wave32) ----------------
// A 16x32 bf16: lane L holds row M=L%16; elems 0..7 = K (L/16)*8+0..7, elems 8..15 = K 16+(L/16)*8+0..7
DEV v16bf frag_a(const bf16* p, int stride) {
  int l = threadIdx.x & 31;
  const bf16* q = p + (l & 15) * stride + ((l >> 4) << 3);
  v8bf lo = *(const v8bf*)q;
  v8bf hi = *(const v8bf*)(q + 16);
  v16bf r;
#pragma unroll
  for (int i = 0; i < 8; ++i) { r[i] = lo[i]; r[i + 8] = hi[i]; }
  return r;
}
// B 32x16 bf16 from row-major Bsrc[n][k]: lane L holds col N=L%16; elems 0..7 = K (L/16)*8+0..7, elems 8..15 = K 16+(L/16)*8+0..7
DEV v16bf frag_b(const bf16* p, int stride) {
  int l = threadIdx.x & 31;
  const bf16* q = p + (l & 15) * stride + ((l >> 4) << 3);
  v8bf lo = *(const v8bf*)q;
  v8bf hi = *(const v8bf*)(q + 16);
  v16bf r;
#pragma unroll
  for (int i = 0; i < 8; ++i) { r[i] = lo[i]; r[i + 8] = hi[i]; }
  return r;
}

DEV v8f wmma_bf16(v16bf a, v16bf b, v8f c) {
  return __builtin_amdgcn_wmma_f32_16x16x32_bf16(false, a, false, b, (short)0, c, false, false);
}

// DPP-based 16-lane reductions (ROW_XMASK butterflies, ROW_SHARE broadcast) — pure VALU
template <int CTRL>
DEV float dpp_mov_f(float v) {
  int r = __builtin_amdgcn_update_dpp(0, __builtin_bit_cast(int, v), CTRL, 0xF, 0xF, true);
  return __builtin_bit_cast(float, r);
}
DEV float rmax16(float v) {
  v = fmaxf(v, dpp_mov_f<0x161>(v));  // ROW_XMASK:1
  v = fmaxf(v, dpp_mov_f<0x162>(v));  // ROW_XMASK:2
  v = fmaxf(v, dpp_mov_f<0x164>(v));  // ROW_XMASK:4
  v = fmaxf(v, dpp_mov_f<0x168>(v));  // ROW_XMASK:8
  return v;
}
DEV float rsum16(float v) {
  v += dpp_mov_f<0x161>(v);
  v += dpp_mov_f<0x162>(v);
  v += dpp_mov_f<0x164>(v);
  v += dpp_mov_f<0x168>(v);
  return v;
}
DEV float bcastlane0(float v) { return dpp_mov_f<0x150>(v); }  // ROW_SHARE:0

// ---------------- kernel 1: per-batch LayerNorm stats ----------------
__global__ __launch_bounds__(256) void k_stats(const float* __restrict__ x,
                                               float* __restrict__ stats) {
  int b = blockIdx.x, tid = threadIdx.x;
  const float* p = x + (size_t)b * CH * IMG * IMG;
  double ds = 0.0, dq = 0.0;
  for (int i = tid; i < CH * IMG * IMG; i += 256) {
    float v = p[i];
    ds += v; dq += (double)v * v;
  }
#pragma unroll
  for (int m = 1; m < 32; m <<= 1) { ds += __shfl_xor(ds, m, 32); dq += __shfl_xor(dq, m, 32); }
  __shared__ double w1[8], w2[8];
  int wv = tid >> 5, ln = tid & 31;
  if (ln == 0) { w1[wv] = ds; w2[wv] = dq; }
  __syncthreads();
  if (tid == 0) {
    double a = 0, q = 0;
#pragma unroll
    for (int i = 0; i < 8; ++i) { a += w1[i]; q += w2[i]; }
    double n = (double)CH * IMG * IMG;
    double mu = a / n;
    double var = q / n - mu * mu;
    stats[b * 2 + 0] = (float)mu;
    stats[b * 2 + 1] = (float)(1.0 / sqrt(var + 1e-5));
  }
}

// ---------------- kernel 2: pixel-unshuffled flash attention ----------------
__global__ __launch_bounds__(256) void k_attn(const float* __restrict__ x,
                                              const float* __restrict__ wqk,
                                              const float* __restrict__ stats,
                                              bf16* __restrict__ a_out) {
  extern __shared__ bf16 sm[];
  bf16* Q    = sm;                 // [1024][QSTR]  xx, later q (bf16)
  bf16* Vt   = sm + ATT_V_OFF;     // [64][VSTR]    V transposed
  bf16* Wt   = sm + ATT_W_OFF;     // [64][WSTR]    wqk transposed
  bf16* Pab  = sm + ATT_P_OFF;     // per-wave 16x40 staging
  bf16* Ones = sm + ATT_ONE_OFF;   // [16][32] row0 = 1, rest 0

  int s = blockIdx.x;
  int b = s >> 4, fi = (s >> 2) & 3, fj = s & 3;
  int tid = threadIdx.x;
  float mu = stats[b * 2 + 0], rs = stats[b * 2 + 1];
  const float* xb = x + (size_t)b * CH * IMG * IMG;

  // stage: pixel-unshuffle gather of x -> Vt (raw) and Q (normalized)
  for (int idx = tid; idx < CH * NPIX; idx += 256) {
    int cc = idx >> 10, n = idx & (NPIX - 1);
    int h = n >> 5, w = n & 31;
    float xv = xb[(cc * IMG + h * FS + fi) * IMG + w * FS + fj];
    Vt[cc * VSTR + n] = (bf16)xv;
    Q[n * QSTR + cc] = (bf16)((xv - mu) * rs);
  }
  for (int idx = tid; idx < CH * EMB; idx += 256) {
    int cc = idx >> 6, e = idx & 63;
    Wt[e * WSTR + cc] = (bf16)wqk[cc * EMB + e];
  }
  for (int idx = tid; idx < 16 * 32; idx += 256)
    Ones[idx] = (bf16)((idx < 32) ? 1.0f : 0.0f);
  __syncthreads();

  int wv = tid >> 5, ln = tid & 31;

  // phase 1: q = normalize(xx @ wqk), overwrite Q rows in place (per-wave owned rows)
  for (int qt = wv; qt < 64; qt += 8) {
    v16bf a0 = frag_a(Q + qt * 16 * QSTR, QSTR);
    v16bf a1 = frag_a(Q + qt * 16 * QSTR + 32, QSTR);
    v8f acc[4];
#pragma unroll
    for (int ec = 0; ec < 4; ++ec) {
      v8f z = {0.f, 0.f, 0.f, 0.f, 0.f, 0.f, 0.f, 0.f};
      z = wmma_bf16(a0, frag_b(Wt + ec * 16 * WSTR, WSTR), z);
      z = wmma_bf16(a1, frag_b(Wt + ec * 16 * WSTR + 32, WSTR), z);
      acc[ec] = z;
    }
    int rbase = qt * 16 + ((ln >> 4) << 3);
    int cb = ln & 15;
#pragma unroll
    for (int r = 0; r < 8; ++r) {
      float sq = 0.f;
#pragma unroll
      for (int ec = 0; ec < 4; ++ec) sq += acc[ec][r] * acc[ec][r];
      sq = rsum16(sq);
      float sc = 1.0f / fmaxf(sqrtf(sq), 1e-12f);
#pragma unroll
      for (int ec = 0; ec < 4; ++ec)
        Q[(rbase + r) * QSTR + ec * 16 + cb] = (bf16)(acc[ec][r] * sc);
    }
  }
  __syncthreads();

  // phase 2: flash attention; l accumulated by ones-column WMMA
  bf16* Pst = Pab + wv * 16 * PSTR;
  v16bf bOnes = frag_b(Ones, 32);
  for (int qt = wv; qt < 64; qt += 8) {
    v16bf aQ0 = frag_a(Q + qt * 16 * QSTR, QSTR);
    v16bf aQ1 = frag_a(Q + qt * 16 * QSTR + 32, QSTR);
    float mrow[8];
    v8f O[4], Ol;
#pragma unroll
    for (int r = 0; r < 8; ++r) mrow[r] = -1e30f;
#pragma unroll
    for (int ec = 0; ec < 4; ++ec) O[ec] = (v8f){0.f, 0.f, 0.f, 0.f, 0.f, 0.f, 0.f, 0.f};
    Ol = (v8f){0.f, 0.f, 0.f, 0.f, 0.f, 0.f, 0.f, 0.f};

    for (int kc = 0; kc < 32; ++kc) {
      int kt0 = kc * 2, kt1 = kc * 2 + 1;
      v8f z0 = {0.f, 0.f, 0.f, 0.f, 0.f, 0.f, 0.f, 0.f};
      z0 = wmma_bf16(aQ0, frag_b(Q + kt0 * 16 * QSTR, QSTR), z0);
      z0 = wmma_bf16(aQ1, frag_b(Q + kt0 * 16 * QSTR + 32, QSTR), z0);
      v8f z1 = {0.f, 0.f, 0.f, 0.f, 0.f, 0.f, 0.f, 0.f};
      z1 = wmma_bf16(aQ0, frag_b(Q + kt1 * 16 * QSTR, QSTR), z1);
      z1 = wmma_bf16(aQ1, frag_b(Q + kt1 * 16 * QSTR + 32, QSTR), z1);

      int rb = (ln >> 4) << 3;
#pragma unroll
      for (int r = 0; r < 8; ++r) {
        float v0 = z0[r] * SOFTMAX_SCL, v1 = z1[r] * SOFTMAX_SCL;  // log2 domain
        float tm = rmax16(fmaxf(v0, v1));
        float mnew = fmaxf(mrow[r], tm);
        float fac = __builtin_amdgcn_exp2f(mrow[r] - mnew);
        mrow[r] = mnew;
        float p0 = __builtin_amdgcn_exp2f(v0 - mnew);
        float p1 = __builtin_amdgcn_exp2f(v1 - mnew);
#pragma unroll
        for (int ec = 0; ec < 4; ++ec) O[ec][r] *= fac;
        Ol[r] *= fac;
        Pst[(rb + r) * PSTR + (ln & 15)] = (bf16)p0;
        Pst[(rb + r) * PSTR + 16 + (ln & 15)] = (bf16)p1;
      }
      v16bf aP = frag_a(Pst, PSTR);
#pragma unroll
      for (int ec = 0; ec < 4; ++ec)
        O[ec] = wmma_bf16(aP, frag_b(Vt + ec * 16 * VSTR + kc * 32, VSTR), O[ec]);
      Ol = wmma_bf16(aP, bOnes, Ol);  // row sums of P land in column 0
    }

    // epilogue: a = x + attn, pixel-shuffle scatter
    int rb = (ln >> 4) << 3;
#pragma unroll
    for (int r = 0; r < 8; ++r) {
      int n = qt * 16 + rb + r;
      int h = n >> 5, w = n & 31;
      float inv = 1.0f / bcastlane0(Ol[r]);
#pragma unroll
      for (int ec = 0; ec < 4; ++ec) {
        int cc = ec * 16 + (ln & 15);
        size_t g = ((size_t)(b * CH + cc) * IMG + h * FS + fi) * IMG + w * FS + fj;
        a_out[g] = (bf16)(x[g] + O[ec][r] * inv);
      }
    }
  }
}

// ---------------- kernel 3: 3x3 dilated(2) conv + ReLU, implicit GEMM ----------------
__global__ __launch_bounds__(256) void k_conv3x3(const bf16* __restrict__ in,
                                                 const float* __restrict__ wt,
                                                 const float* __restrict__ bias,
                                                 bf16* __restrict__ out) {
  extern __shared__ bf16 sm[];
  bf16* P  = sm;              // [36*36][CSTR]  halo patch, [pixel][ci]
  bf16* Wc = sm + CV_W_OFF;   // [64][WCSTR]    weights, [co][(kh*3+kw)*64+ci]
  int blk = blockIdx.x;
  int b = blk >> 4, tile = blk & 15;
  int ty0 = (tile >> 2) * 32, tx0 = (tile & 3) * 32;
  int tid = threadIdx.x;
  const bf16* inb = in + (size_t)b * CH * IMG * IMG;

  for (int idx = tid; idx < PATCH * PATCH * CH; idx += 256) {
    int ci = idx / (PATCH * PATCH);
    int p = idx - ci * (PATCH * PATCH);
    int py = p / PATCH, px = p - py * PATCH;
    int ih = ty0 + py - 2, iw = tx0 + px - 2;
    bf16 v = (bf16)0.0f;
    if (ih >= 0 && ih < IMG && iw >= 0 && iw < IMG) v = inb[(ci * IMG + ih) * IMG + iw];
    P[p * CSTR + ci] = v;
  }
  for (int idx = tid; idx < CH * 576; idx += 256) {
    int co = idx / 576, r = idx - co * 576;
    int tap = r >> 6, ci = r & 63;
    int kh = tap / 3, kw = tap - kh * 3;
    Wc[co * WCSTR + r] = (bf16)wt[((co * CH + ci) * 3 + kh) * 3 + kw];
  }
  __syncthreads();

  int wv = tid >> 5, ln = tid & 31;
  for (int mt = wv; mt < 64; mt += 8) {
    int m0 = mt * 16;
    int ty = m0 >> 5, txb = m0 & 31;
    v8f acc[4];
#pragma unroll
    for (int ec = 0; ec < 4; ++ec) acc[ec] = (v8f){0.f, 0.f, 0.f, 0.f, 0.f, 0.f, 0.f, 0.f};
#pragma unroll
    for (int tap = 0; tap < 9; ++tap) {
      int kh = tap / 3, kw = tap - kh * 3;
      const bf16* abase = P + ((ty + 2 * kh) * PATCH + txb + 2 * kw) * CSTR;
#pragma unroll
      for (int cih = 0; cih < 2; ++cih) {
        v16bf a = frag_a(abase + cih * 32, CSTR);
#pragma unroll
        for (int ec = 0; ec < 4; ++ec)
          acc[ec] = wmma_bf16(a, frag_b(Wc + ec * 16 * WCSTR + tap * 64 + cih * 32, WCSTR), acc[ec]);
      }
    }
    int rb = (ln >> 4) << 3;
#pragma unroll
    for (int ec = 0; ec < 4; ++ec) {
      int co = ec * 16 + (ln & 15);
      float bs = bias[co];
      v8bf pk;
#pragma unroll
      for (int r = 0; r < 8; ++r) pk[r] = (bf16)fmaxf(acc[ec][r] + bs, 0.0f);
      // 8 consecutive pixels, same co -> one 16B store
      *(v8bf*)&out[((size_t)(b * CH + co) * IMG + ty0 + ty) * IMG + tx0 + txb + rb] = pk;
    }
  }
}

// ---------------- kernel 4: 1x1 conv + bias + residual ----------------
__global__ __launch_bounds__(256) void k_conv1x1res(const bf16* __restrict__ y2,
                                                    const float* __restrict__ w3,
                                                    const float* __restrict__ b3,
                                                    const float* __restrict__ x,
                                                    float* __restrict__ out) {
  extern __shared__ bf16 sm[];
  bf16* Y  = sm;              // [1024][QSTR]  [pixel][ci]
  bf16* W3 = sm + K3_W_OFF;   // [64][WSTR]    [co][ci]
  int blk = blockIdx.x;
  int b = blk >> 4, tile = blk & 15;
  int ty0 = (tile >> 2) * 32, tx0 = (tile & 3) * 32;
  int tid = threadIdx.x;
  const bf16* yb = y2 + (size_t)b * CH * IMG * IMG;

  for (int idx = tid; idx < CH * NPIX; idx += 256) {
    int ci = idx >> 10, p = idx & 1023;
    int py = p >> 5, px = p & 31;
    Y[p * QSTR + ci] = yb[(ci * IMG + ty0 + py) * IMG + tx0 + px];
  }
  for (int idx = tid; idx < CH * CH; idx += 256) {
    int co = idx >> 6, ci = idx & 63;
    W3[co * WSTR + ci] = (bf16)w3[co * CH + ci];
  }
  __syncthreads();

  int wv = tid >> 5, ln = tid & 31;
  for (int mt = wv; mt < 64; mt += 8) {
    v16bf a0 = frag_a(Y + mt * 16 * QSTR, QSTR);
    v16bf a1 = frag_a(Y + mt * 16 * QSTR + 32, QSTR);
    int rb = (ln >> 4) << 3;
    int p0 = mt * 16 + rb;
    int py = p0 >> 5, px = p0 & 31;
#pragma unroll
    for (int ec = 0; ec < 4; ++ec) {
      v8f z = {0.f, 0.f, 0.f, 0.f, 0.f, 0.f, 0.f, 0.f};
      z = wmma_bf16(a0, frag_b(W3 + ec * 16 * WSTR, WSTR), z);
      z = wmma_bf16(a1, frag_b(W3 + ec * 16 * WSTR + 32, WSTR), z);
      int co = ec * 16 + (ln & 15);
      float bs = b3[co];
      size_t g = ((size_t)(b * CH + co) * IMG + ty0 + py) * IMG + tx0 + px;
      float4 o0 = *(const float4*)(x + g);
      float4 o1 = *(const float4*)(x + g + 4);
      o0.x += z[0] + bs; o0.y += z[1] + bs; o0.z += z[2] + bs; o0.w += z[3] + bs;
      o1.x += z[4] + bs; o1.y += z[5] + bs; o1.z += z[6] + bs; o1.w += z[7] + bs;
      *(float4*)(out + g) = o0;
      *(float4*)(out + g + 4) = o1;
    }
  }
}

// ---------------- host ----------------
extern "C" void kernel_launch(void* const* d_in, const int* in_sizes, int n_in,
                              void* d_out, int out_size, void* d_ws, size_t ws_size,
                              hipStream_t stream) {
  (void)in_sizes; (void)n_in; (void)out_size; (void)ws_size;
  const float* x   = (const float*)d_in[0];
  const float* wqk = (const float*)d_in[1];
  const float* c1w = (const float*)d_in[2];
  const float* c1b = (const float*)d_in[3];
  const float* c2w = (const float*)d_in[4];
  const float* c2b = (const float*)d_in[5];
  const float* c3w = (const float*)d_in[6];
  const float* c3b = (const float*)d_in[7];
  float* out = (float*)d_out;

  char* ws = (char*)d_ws;
  float* stats = (float*)ws;
  size_t tsz = (size_t)BATCH * CH * IMG * IMG * sizeof(bf16);  // 16.8 MB
  bf16* a_buf  = (bf16*)(ws + 256);         // attention output, reused as y2
  bf16* y1_buf = (bf16*)(ws + 256 + tsz);

  hipFuncSetAttribute(reinterpret_cast<const void*>(k_attn),
                      hipFuncAttributeMaxDynamicSharedMemorySize, (int)ATT_LDS_BYTES);
  hipFuncSetAttribute(reinterpret_cast<const void*>(k_conv3x3),
                      hipFuncAttributeMaxDynamicSharedMemorySize, (int)CV_LDS_BYTES);
  hipFuncSetAttribute(reinterpret_cast<const void*>(k_conv1x1res),
                      hipFuncAttributeMaxDynamicSharedMemorySize, (int)K3_LDS_BYTES);

  k_stats<<<BATCH, 256, 0, stream>>>(x, stats);
  k_attn<<<128, 256, ATT_LDS_BYTES, stream>>>(x, wqk, stats, a_buf);
  k_conv3x3<<<128, 256, CV_LDS_BYTES, stream>>>(a_buf, c1w, c1b, y1_buf);
  k_conv3x3<<<128, 256, CV_LDS_BYTES, stream>>>(y1_buf, c2w, c2b, a_buf);  // y2 -> a_buf
  k_conv1x1res<<<128, 256, K3_LDS_BYTES, stream>>>(a_buf, c3w, c3b, x, out);
}
